// CRF_71794673320660
// MI455X (gfx1250) — matrix operand order repaired and checked by reading, not compiled, not run on hardware
//
#include <hip/hip_runtime.h>
#include <math.h>

typedef __attribute__((ext_vector_type(16))) _Float16 v16h;
typedef __attribute__((ext_vector_type(8)))  _Float16 h8;
typedef __attribute__((ext_vector_type(8)))  float    v8f;

#define Bn 128
#define Sn 512
#define Cn 128

#define LOG2E 1.4426950408889634f
#define LN2   0.6931471805599453f

// ---------------------------------------------------------------------------
// Partition function: 8 WGs x 16 batch rows. 8 wave32 per WG, each wave owns a
// 16-column tile of next-alpha. Per step: one 16x16x128 GEMM in the
// exp-rescaled domain via 4 chained v_wmma_f32_16x16x32_f16, with the
// transition matrix (exp(T - colmax), f16) pinned in VGPRs for all 512 steps.
// ---------------------------------------------------------------------------
__global__ __launch_bounds__(256) void crf_partition_kernel(
    const float* __restrict__ emis,          // [B,S,C] f32
    const unsigned char* __restrict__ mask,  // [B,S] bool (1 byte)
    const float* __restrict__ trans,         // [C,C] f32
    const float* __restrict__ start_t,       // [C]
    const float* __restrict__ end_t,         // [C]
    float* __restrict__ part)                // [B] out
{
  __shared__ _Float16 Abuf[16 * Cn];   // A operand staging: 16 rows x 128 K, f16
  __shared__ float    maxbuf[8][16];   // per-wave per-row partial max
  __shared__ float    rowmax16[16];    // combined per-row max (final stage)
  __shared__ float    sumbuf[8][16];   // final logsumexp partial sums

  const int tid  = threadIdx.x;
  const int wave = tid >> 5;
  const int lane = tid & 31;
  const int hi   = (lane >= 16) ? 1 : 0;
  const int n    = lane & 15;          // column within wave tile / A row
  const int col  = wave * 16 + n;      // global destination column j
  const int mb   = hi * 8;             // C/D row offset for this lane half
  const int rowsBase = blockIdx.x * 16;

  // Per-column max of transitions (rescale so exp(T - cmax) in (0,1]).
  float cmax = -INFINITY;
  for (int i = 0; i < Cn; ++i) cmax = fmaxf(cmax, trans[i * Cn + col]);

  // B operand (32x16 f16 per K-block) resident in VGPRs for the whole scan.
  // B layout: lanes 0-15 hold K=0..15, lanes 16-31 hold K=16..31; lane%16 = N.
  v16h Breg[4];
  #pragma unroll
  for (int kb = 0; kb < 4; ++kb) {
    #pragma unroll
    for (int j = 0; j < 16; ++j) {
      int i = kb * 32 + hi * 16 + j;   // transition source row (K index)
      Breg[kb][j] =
          (_Float16)__builtin_amdgcn_exp2f((trans[i * Cn + col] - cmax) * LOG2E);
    }
  }

  const float endv = end_t[col];

  // alpha in the f32 C/D layout: reg r <-> local row (mb+r), column col.
  float alpha[8];
  #pragma unroll
  for (int r = 0; r < 8; ++r) {
    int row = rowsBase + mb + r;
    alpha[r] = start_t[col] + emis[((size_t)row * Sn + 0) * Cn + col];
  }

  for (int t = 1; t < Sn; ++t) {
    // ---- issue this step's emission/mask loads early (latency hidden
    //      under the reduction + WMMA below)
    float e[8];
    int   mk[8];
    #pragma unroll
    for (int r = 0; r < 8; ++r) {
      int row = rowsBase + mb + r;
      e[r]  = emis[((size_t)row * Sn + t) * Cn + col];
      mk[r] = mask[(size_t)row * Sn + t];
    }

    // ---- per-row max across the wave's 16 columns (xor stays in 16-lane half)
    float lm[8];
    #pragma unroll
    for (int r = 0; r < 8; ++r) lm[r] = alpha[r];
    #pragma unroll
    for (int s = 1; s < 16; s <<= 1) {
      #pragma unroll
      for (int r = 0; r < 8; ++r) lm[r] = fmaxf(lm[r], __shfl_xor(lm[r], s));
    }
    if (lane == 0) {
      #pragma unroll
      for (int r = 0; r < 8; ++r) maxbuf[wave][r] = lm[r];
    }
    if (lane == 16) {
      #pragma unroll
      for (int r = 0; r < 8; ++r) maxbuf[wave][8 + r] = lm[r];
    }
    __syncthreads();

    // Every lane combines the cross-wave max for row (lane&15), then the 8
    // per-row maxima this lane needs are fetched by intra-wave shuffles --
    // avoids a second barrier + LDS broadcast on the critical path.
    float comb = maxbuf[0][n];
    #pragma unroll
    for (int w = 1; w < 8; ++w) comb = fmaxf(comb, maxbuf[w][n]);
    float rmax[8];
    #pragma unroll
    for (int r = 0; r < 8; ++r) rmax[r] = __shfl(comb, mb + r);

    // ---- stage A = exp(alpha - rowmax) as f16 into LDS (row-major 16x128)
    #pragma unroll
    for (int r = 0; r < 8; ++r)
      Abuf[(mb + r) * Cn + col] =
          (_Float16)__builtin_amdgcn_exp2f((alpha[r] - rmax[r]) * LOG2E);
    __syncthreads();

    // ---- 4 chained WMMAs over K. A layout: row = lane%16; halves 0-7 cover
    // K = kb*32 + (hi?16:0) .. +7, halves 8-15 cover K = +8 .. +15.
    v8f acc = {0.f, 0.f, 0.f, 0.f, 0.f, 0.f, 0.f, 0.f};
    #pragma unroll
    for (int kb = 0; kb < 4; ++kb) {
      const int kLo = kb * 32 + hi * 16;
      h8 lo  = *(const h8*)(&Abuf[n * Cn + kLo]);
      h8 hi8 = *(const h8*)(&Abuf[n * Cn + kLo + 8]);
      v16h a;
      #pragma unroll
      for (int j = 0; j < 8; ++j) { a[j] = lo[j]; a[8 + j] = hi8[j]; }
      acc = __builtin_amdgcn_wmma_f32_16x16x32_f16(
          false, a, false, Breg[kb], (short)0, acc, false, false);
    }

    // ---- back to log domain + emissions; branchless masked select
    float na[8];
    #pragma unroll
    for (int r = 0; r < 8; ++r)
      na[r] = LN2 * __builtin_amdgcn_logf(acc[r]) + (rmax[r] + cmax + e[r]);
    #pragma unroll
    for (int r = 0; r < 8; ++r)
      alpha[r] = mk[r] ? na[r] : alpha[r];
  }

  // ---- final logsumexp over columns of (alpha + end_t)
  float v[8];
  #pragma unroll
  for (int r = 0; r < 8; ++r) v[r] = alpha[r] + endv;

  float lm[8];
  #pragma unroll
  for (int r = 0; r < 8; ++r) lm[r] = v[r];
  #pragma unroll
  for (int s = 1; s < 16; s <<= 1) {
    #pragma unroll
    for (int r = 0; r < 8; ++r) lm[r] = fmaxf(lm[r], __shfl_xor(lm[r], s));
  }
  if (lane == 0) {
    #pragma unroll
    for (int r = 0; r < 8; ++r) maxbuf[wave][r] = lm[r];
  }
  if (lane == 16) {
    #pragma unroll
    for (int r = 0; r < 8; ++r) maxbuf[wave][8 + r] = lm[r];
  }
  __syncthreads();
  if (tid < 16) {
    float m = maxbuf[0][tid];
    #pragma unroll
    for (int w = 1; w < 8; ++w) m = fmaxf(m, maxbuf[w][tid]);
    rowmax16[tid] = m;
  }
  __syncthreads();

  float se[8];
  #pragma unroll
  for (int r = 0; r < 8; ++r)
    se[r] = __builtin_amdgcn_exp2f((v[r] - rowmax16[mb + r]) * LOG2E);
  #pragma unroll
  for (int s = 1; s < 16; s <<= 1) {
    #pragma unroll
    for (int r = 0; r < 8; ++r) se[r] += __shfl_xor(se[r], s);
  }
  if (lane == 0) {
    #pragma unroll
    for (int r = 0; r < 8; ++r) sumbuf[wave][r] = se[r];
  }
  if (lane == 16) {
    #pragma unroll
    for (int r = 0; r < 8; ++r) sumbuf[wave][8 + r] = se[r];
  }
  __syncthreads();
  if (tid < 16) {
    float tot = sumbuf[0][tid];
    #pragma unroll
    for (int w = 1; w < 8; ++w) tot += sumbuf[w][tid];
    part[rowsBase + tid] = LN2 * __builtin_amdgcn_logf(tot) + rowmax16[tid];
  }
}

// ---------------------------------------------------------------------------
// Gold score: one block per batch element, threads strided over sequence.
// ---------------------------------------------------------------------------
__global__ __launch_bounds__(256) void crf_gold_kernel(
    const float* __restrict__ emis, const int* __restrict__ tags,
    const unsigned char* __restrict__ mask, const float* __restrict__ trans,
    const float* __restrict__ start_t, const float* __restrict__ end_t,
    float* __restrict__ gold)
{
  __shared__ float sred[256];
  __shared__ int   scnt[256];
  const int b   = blockIdx.x;
  const int tid = threadIdx.x;

  float acc = 0.f;
  int   cnt = 0;
  for (int s = tid; s < Sn; s += 256) {
    int mk = mask[(size_t)b * Sn + s] ? 1 : 0;
    cnt += mk;
    if (s > 0 && mk) {
      int tc = tags[b * Sn + s];
      int tp = tags[b * Sn + s - 1];
      acc += emis[((size_t)b * Sn + s) * Cn + tc] + trans[tp * Cn + tc];
    }
  }
  sred[tid] = acc;
  scnt[tid] = cnt;
  __syncthreads();
  for (int off = 128; off > 0; off >>= 1) {
    if (tid < off) {
      sred[tid] += sred[tid + off];
      scnt[tid] += scnt[tid + off];
    }
    __syncthreads();
  }
  if (tid == 0) {
    int   t0 = tags[b * Sn];
    float sc = start_t[t0] + emis[((size_t)b * Sn) * Cn + t0] + sred[0];
    int   last = scnt[0] - 1;
    int   tl = tags[b * Sn + last];
    gold[b] = sc + end_t[tl];
  }
}

// ---------------------------------------------------------------------------
// out[0] = mean(part - gold) == -(gold - part).mean()
// ---------------------------------------------------------------------------
__global__ __launch_bounds__(128) void crf_finalize_kernel(
    const float* __restrict__ gold, const float* __restrict__ part,
    float* __restrict__ out)
{
  __shared__ float red[128];
  const int tid = threadIdx.x;
  red[tid] = part[tid] - gold[tid];
  __syncthreads();
  for (int off = 64; off > 0; off >>= 1) {
    if (tid < off) red[tid] += red[tid + off];
    __syncthreads();
  }
  if (tid == 0) out[0] = red[0] / (float)Bn;
}

extern "C" void kernel_launch(void* const* d_in, const int* in_sizes, int n_in,
                              void* d_out, int out_size, void* d_ws, size_t ws_size,
                              hipStream_t stream) {
  const float*         emissions = (const float*)d_in[0];
  const int*           tags      = (const int*)d_in[1];           // int32 on device
  const unsigned char* mask      = (const unsigned char*)d_in[2]; // bool = 1 byte
  const float*         trans     = (const float*)d_in[3];
  const float*         start_t   = (const float*)d_in[4];
  const float*         end_t     = (const float*)d_in[5];
  float* out  = (float*)d_out;
  float* gold = (float*)d_ws;        // [128]
  float* part = gold + Bn;           // [128]

  crf_gold_kernel<<<Bn, 256, 0, stream>>>(emissions, tags, mask, trans,
                                          start_t, end_t, gold);
  crf_partition_kernel<<<Bn / 16, 256, 0, stream>>>(emissions, mask, trans,
                                                    start_t, end_t, part);
  crf_finalize_kernel<<<1, 128, 0, stream>>>(gold, part, out);
}
